// VitBlock_13993003451072
// MI455X (gfx1250) — matrix operand order repaired and checked by reading, not compiled, not run on hardware
//
#include <hip/hip_runtime.h>
#include <math.h>

// ---------------- problem constants ----------------
#define BB   32
#define SS   256
#define DD   768
#define HH   12
#define HD   64
#define MLPD 3072
#define ROWS (BB * SS)          // 8192

// ---------------- types ----------------
typedef __attribute__((ext_vector_type(16))) __bf16 v16bf;
typedef __attribute__((ext_vector_type(8)))  float  v8f;

struct Frag { union { unsigned int u[8]; v16bf v; }; };

// ---------------- small helpers ----------------
__device__ __forceinline__ unsigned short f2bf(float f) {
    __bf16 h = (__bf16)f;                       // native v_cvt, RNE
    return __builtin_bit_cast(unsigned short, h);
}
__device__ __forceinline__ unsigned int pack2(float lo, float hi) {
    return (unsigned int)f2bf(lo) | ((unsigned int)f2bf(hi) << 16);
}
__device__ __forceinline__ v8f splat8(float v) {
    v8f r;
#pragma unroll
    for (int i = 0; i < 8; ++i) r[i] = v;
    return r;
}

// A-fragment: 16x32 bf16, M = lane%16, K-base = (lane/16)*8
__device__ __forceinline__ Frag load_frag_a(const unsigned short* A, int ld,
                                            int row0, int k0, int lane) {
    Frag f;
    int m  = lane & 15;
    int kb = (lane >> 4) << 3;
    const unsigned short* p = A + (size_t)(row0 + m) * ld + k0;
#pragma unroll
    for (int j = 0; j < 8; ++j) {
        int k = ((j >> 2) << 4) + kb + ((j & 3) << 1);
        f.u[j] = *reinterpret_cast<const unsigned int*>(p + k);
    }
    return f;
}

// B-fragment: 32x16 bf16 from a row-major [N][K] matrix (K contiguous).
// N = lane%16, K-base = (lane/16)*8
__device__ __forceinline__ Frag load_frag_b(const unsigned short* Bm, int ld,
                                            int n0, int k0, int lane) {
    Frag f;
    int n  = lane & 15;
    int kb = (lane >> 4) << 3;
    const unsigned short* p = Bm + (size_t)(n0 + n) * ld + k0;
#pragma unroll
    for (int j = 0; j < 8; ++j)
        f.u[j] = *reinterpret_cast<const unsigned int*>(p + ((j >> 2) << 4) + kb + ((j & 3) << 1));
    return f;
}

__device__ __forceinline__ v8f wmma_bf16(const Frag& a, const Frag& b, v8f c) {
    return __builtin_amdgcn_wmma_f32_16x16x32_bf16(
        false, a.v, false, b.v, (short)0, c, false, false);
}

// ---------------- CDNA5 async copy (global -> LDS, ASYNCcnt) ----------------
typedef int v4i_g __attribute__((vector_size(16)));
typedef __attribute__((address_space(1))) v4i_g* as1_v4i;
typedef __attribute__((address_space(3))) v4i_g* as3_v4i;

__device__ __forceinline__ void async_ld16(const unsigned short* g, unsigned short* l) {
#if __has_builtin(__builtin_amdgcn_global_load_async_to_lds_b128)
    __builtin_amdgcn_global_load_async_to_lds_b128((as1_v4i)g, (as3_v4i)l, 0, 0);
#else
    unsigned loff = (unsigned)(uintptr_t)(__attribute__((address_space(3))) unsigned short*)l;
    asm volatile("global_load_async_to_lds_b128 %0, %1, off"
                 :: "v"(loff), "v"(g) : "memory");
#endif
}

#if __has_builtin(__builtin_amdgcn_s_wait_asynccnt)
#define WAIT_ASYNC(n) __builtin_amdgcn_s_wait_asynccnt(n)
#else
#define WAIT_ASYNC(n) asm volatile("s_wait_asynccnt " #n ::: "memory")
#endif

// ---------------- kernels ----------------

// fp32 -> bf16 elementwise
__global__ void cvt_bf16_kernel(const float* __restrict__ in,
                                unsigned short* __restrict__ out, int n) {
    int i = blockIdx.x * 256 + threadIdx.x;
    if (i < n) out[i] = f2bf(in[i]);
}

// LayerNorm over D=768, one row per 256-thread block, bf16 output
__global__ void ln_kernel(const float* __restrict__ x,
                          const float* __restrict__ g,
                          const float* __restrict__ be,
                          unsigned short* __restrict__ out) {
    const size_t base = (size_t)blockIdx.x * DD;
    const int t = threadIdx.x;
    float v0 = x[base + t];
    float v1 = x[base + t + 256];
    float v2 = x[base + t + 512];
    float s  = v0 + v1 + v2;
    float s2 = v0 * v0 + v1 * v1 + v2 * v2;
#pragma unroll
    for (int m = 16; m > 0; m >>= 1) {
        s  += __shfl_xor(s,  m, 32);
        s2 += __shfl_xor(s2, m, 32);
    }
    __shared__ float sm[8], sm2[8];
    if ((t & 31) == 0) { sm[t >> 5] = s; sm2[t >> 5] = s2; }
    __syncthreads();
    float S = 0.f, S2 = 0.f;
#pragma unroll
    for (int w = 0; w < 8; ++w) { S += sm[w]; S2 += sm2[w]; }
    const float mean = S * (1.0f / DD);
    const float var  = S2 * (1.0f / DD) - mean * mean;
    const float rstd = rsqrtf(var + 1e-5f);
    out[base + t]       = f2bf((v0 - mean) * rstd * g[t]       + be[t]);
    out[base + t + 256] = f2bf((v1 - mean) * rstd * g[t + 256] + be[t + 256]);
    out[base + t + 512] = f2bf((v2 - mean) * rstd * g[t + 512] + be[t + 512]);
}

// QKV projection: one block per (b,h), 8 waves, 192 16x16 tiles total.
// q,k stored [bh][s][e] bf16; v stored transposed [bh][e][t] bf16.
__global__ void qkv_kernel(const unsigned short* __restrict__ hb,
                           const unsigned short* __restrict__ Wqb, const float* __restrict__ bq,
                           const unsigned short* __restrict__ Wkb, const float* __restrict__ bk,
                           const unsigned short* __restrict__ Wvb, const float* __restrict__ bv,
                           unsigned short* __restrict__ q,
                           unsigned short* __restrict__ k,
                           unsigned short* __restrict__ vT) {
    const int bh = blockIdx.x;
    const int b  = bh / HH;
    const int h  = bh % HH;
    const int lane = threadIdx.x & 31;
    const int wave = threadIdx.x >> 5;

    const unsigned short* Abase = hb + (size_t)b * SS * DD;

    for (int t = wave; t < 192; t += 8) {                      // uniform: 24 iters per wave
        const int mat = t / 64;
        const int rem = t % 64;
        const int mt  = rem >> 2;
        const int nt  = rem & 3;

        const unsigned short* Wb;
        const float* bias;
        if (mat == 0)      { Wb = Wqb; bias = bq; }
        else if (mat == 1) { Wb = Wkb; bias = bk; }
        else               { Wb = Wvb; bias = bv; }
        Wb   += (size_t)h * HD * HD;
        bias += h * HD;

        v8f acc = splat8(bias[nt * 16 + (lane & 15)]);

        Frag a0 = load_frag_a(Abase, DD, mt * 16, h * HD + 0,  lane);
        Frag b0 = load_frag_b(Wb,    HD, nt * 16, 0,           lane);
        acc = wmma_bf16(a0, b0, acc);
        Frag a1 = load_frag_a(Abase, DD, mt * 16, h * HD + 32, lane);
        Frag b1 = load_frag_b(Wb,    HD, nt * 16, 32,          lane);
        acc = wmma_bf16(a1, b1, acc);

        const int n  = lane & 15;
        const int mb = (lane >> 4) << 3;
        if (mat < 2) {
            unsigned short* dst = (mat == 0 ? q : k) + (size_t)bh * SS * HD;
#pragma unroll
            for (int j = 0; j < 8; ++j)
                dst[(size_t)(mt * 16 + mb + j) * HD + nt * 16 + n] = f2bf(acc[j]);
        } else {
            size_t base = (size_t)bh * HD * SS + (size_t)(nt * 16 + n) * SS + mt * 16 + mb;
            uint4 val;
            val.x = pack2(acc[0], acc[1]);
            val.y = pack2(acc[2], acc[3]);
            val.z = pack2(acc[4], acc[5]);
            val.w = pack2(acc[6], acc[7]);
            *reinterpret_cast<uint4*>(vT + base) = val;
        }
    }
}

// Attention: 128 threads = 4 waves; each wave owns one 16-row tile of one (b,h).
__global__ void attn_kernel(const unsigned short* __restrict__ q,
                            const unsigned short* __restrict__ k,
                            const unsigned short* __restrict__ vT,
                            const float* __restrict__ x,
                            float* __restrict__ out1) {
    __shared__ __align__(16) unsigned short lds[4 * 16 * SS];   // 32 KB
    const int lane = threadIdx.x & 31;
    const int wave = threadIdx.x >> 5;
    const int g    = blockIdx.x * 4 + wave;
    const int bh   = g >> 4;
    const int st   = g & 15;
    const int b    = bh / HH;
    const int h    = bh % HH;

    const unsigned short* qp = q  + (size_t)bh * SS * HD;
    const unsigned short* kp = k  + (size_t)bh * SS * HD;
    const unsigned short* vp = vT + (size_t)bh * HD * SS;

    Frag a0 = load_frag_a(qp, HD, st * 16, 0,  lane);
    Frag a1 = load_frag_a(qp, HD, st * 16, 32, lane);
    v8f acc[16];
#pragma unroll
    for (int nt = 0; nt < 16; ++nt) acc[nt] = splat8(0.f);
#pragma unroll
    for (int nt = 0; nt < 16; ++nt) {
        Frag b0 = load_frag_b(kp, HD, nt * 16, 0,  lane);
        acc[nt] = wmma_bf16(a0, b0, acc[nt]);
        Frag b1 = load_frag_b(kp, HD, nt * 16, 32, lane);
        acc[nt] = wmma_bf16(a1, b1, acc[nt]);
    }

    const float sc = 0.125f;   // 1/sqrt(64)
    float rsum[8];
#pragma unroll
    for (int j = 0; j < 8; ++j) {
        float m = -3.4e38f;
#pragma unroll
        for (int nt = 0; nt < 16; ++nt) {
            acc[nt][j] *= sc;
            m = fmaxf(m, acc[nt][j]);
        }
#pragma unroll
        for (int msk = 1; msk < 16; msk <<= 1)
            m = fmaxf(m, __shfl_xor(m, msk, 32));
        float s = 0.f;
#pragma unroll
        for (int nt = 0; nt < 16; ++nt) {
            float e = __expf(acc[nt][j] - m);
            acc[nt][j] = e;
            s += e;
        }
#pragma unroll
        for (int msk = 1; msk < 16; msk <<= 1)
            s += __shfl_xor(s, msk, 32);
        rsum[j] = 1.0f / s;
    }

    unsigned short* ldsw = lds + wave * (16 * SS);
    const int n  = lane & 15;
    const int mb = (lane >> 4) << 3;
#pragma unroll
    for (int nt = 0; nt < 16; ++nt)
#pragma unroll
        for (int j = 0; j < 8; ++j)
            ldsw[(mb + j) * SS + nt * 16 + n] = f2bf(acc[nt][j] * rsum[j]);
    __syncthreads();

    v8f acc2[4];
#pragma unroll
    for (int nt = 0; nt < 4; ++nt) acc2[nt] = splat8(0.f);
#pragma unroll
    for (int kt = 0; kt < 8; ++kt) {
        Frag af = load_frag_a(ldsw, SS, 0, kt * 32, lane);
#pragma unroll
        for (int nt = 0; nt < 4; ++nt) {
            Frag bf_ = load_frag_b(vp, SS, nt * 16, kt * 32, lane);
            acc2[nt] = wmma_bf16(af, bf_, acc2[nt]);
        }
    }

#pragma unroll
    for (int nt = 0; nt < 4; ++nt) {
        const int d = h * HD + nt * 16 + n;
#pragma unroll
        for (int j = 0; j < 8; ++j) {
            const int s   = st * 16 + mb + j;
            const size_t idx = ((size_t)b * SS + s) * DD + d;
            out1[idx] = acc2[nt][j] + x[idx];
        }
    }
}

// ---------------- async-staged MLP GEMM core ----------------
// Block: 256 threads / 8 waves, block tile 64x256, K-step 32, double-buffered LDS.
// Per stage per thread: 1 A-chunk + 4 B-chunks (16B async copies) -> ASYNCcnt = 5.
template <int LDK, int NK, int GELU>
__device__ __forceinline__ void mlp_core(const unsigned short* __restrict__ A,
                                         const unsigned short* __restrict__ W,
                                         const float* __restrict__ bias,
                                         const float* __restrict__ res,
                                         float* __restrict__ outf,
                                         unsigned short* __restrict__ outb,
                                         int outld) {
    __shared__ __align__(16) unsigned short sA[2][64 * 32];    // 8 KB
    __shared__ __align__(16) unsigned short sB[2][256 * 32];   // 32 KB
    const int lane = threadIdx.x & 31;
    const int w    = threadIdx.x >> 5;
    const int bx   = blockIdx.x % (ROWS / 64);
    const int by   = blockIdx.x / (ROWS / 64);
    const int rowblk = bx * 64;
    const int colblk = by * 256;
    const int row0w  = (w & 1) * 32;        // within block tile
    const int col0w  = (w >> 1) * 64;

    const int t  = threadIdx.x;
    const int ra = t >> 2,  ca = (t & 3) * 8;                  // A: 64 rows x 4 chunks

    v8f acc[2][4];
#pragma unroll
    for (int nt = 0; nt < 4; ++nt) {
        float bv = bias[colblk + col0w + nt * 16 + (lane & 15)];
        acc[0][nt] = splat8(bv);
        acc[1][nt] = splat8(bv);
    }

    auto issue = [&](int kt, int bufi) {
        async_ld16(A + (size_t)(rowblk + ra) * LDK + kt * 32 + ca,
                   &sA[bufi][ra * 32 + ca]);
#pragma unroll
        for (int i = 0; i < 4; ++i) {
            int id = t + 256 * i;
            int rb = id >> 2, cb = (id & 3) * 8;               // B: 256 rows x 4 chunks
            async_ld16(W + (size_t)(colblk + rb) * LDK + kt * 32 + cb,
                       &sB[bufi][rb * 32 + cb]);
        }
    };

    issue(0, 0);
    for (int kt = 0; kt < NK; ++kt) {
        const int bufi = kt & 1;
        if (kt + 1 < NK) {
            issue(kt + 1, bufi ^ 1);
            WAIT_ASYNC(5);                  // oldest stage done (in-order)
        } else {
            WAIT_ASYNC(0);
        }
        __syncthreads();
        Frag a0 = load_frag_a(sA[bufi], 32, row0w,      0, lane);
        Frag a1 = load_frag_a(sA[bufi], 32, row0w + 16, 0, lane);
#pragma unroll
        for (int nt = 0; nt < 4; ++nt) {
            Frag bfr = load_frag_b(sB[bufi], 32, col0w + nt * 16, 0, lane);
            acc[0][nt] = wmma_bf16(a0, bfr, acc[0][nt]);
            acc[1][nt] = wmma_bf16(a1, bfr, acc[1][nt]);
        }
        __syncthreads();                    // buffer free for next issue
    }

    const int n  = lane & 15;
    const int mb = (lane >> 4) << 3;
#pragma unroll
    for (int mi = 0; mi < 2; ++mi)
#pragma unroll
        for (int nt = 0; nt < 4; ++nt)
#pragma unroll
            for (int j = 0; j < 8; ++j) {
                const int r = rowblk + row0w + mi * 16 + mb + j;
                const int c = colblk + col0w + nt * 16 + n;
                float v = acc[mi][nt][j];
                if (GELU) {
                    float gel = 0.5f * v * (1.0f + erff(v * 0.70710678118f));
                    outb[(size_t)r * outld + c] = f2bf(gel);
                } else {
                    const size_t idx = (size_t)r * outld + c;
                    outf[idx] = v + res[idx];
                }
            }
}

__global__ void mlp1_kernel(const unsigned short* __restrict__ A,
                            const unsigned short* __restrict__ W,
                            const float* __restrict__ bias,
                            unsigned short* __restrict__ out) {
    mlp_core<DD, DD / 32, 1>(A, W, bias, nullptr, nullptr, out, MLPD);
}

__global__ void mlp2_kernel(const unsigned short* __restrict__ A,
                            const unsigned short* __restrict__ W,
                            const float* __restrict__ bias,
                            const float* __restrict__ res,
                            float* __restrict__ out) {
    mlp_core<MLPD, MLPD / 32, 0>(A, W, bias, res, out, nullptr, DD);
}

// ---------------- host launcher ----------------
static inline size_t align256(size_t x) { return (x + 255) & ~(size_t)255; }

extern "C" void kernel_launch(void* const* d_in, const int* in_sizes, int n_in,
                              void* d_out, int out_size, void* d_ws, size_t ws_size,
                              hipStream_t stream) {
    const float* x     = (const float*)d_in[0];
    const float* g1    = (const float*)d_in[1];
    const float* beta1 = (const float*)d_in[2];
    const float* Wq    = (const float*)d_in[3];
    const float* bq    = (const float*)d_in[4];
    const float* Wk    = (const float*)d_in[5];
    const float* bk    = (const float*)d_in[6];
    const float* Wv    = (const float*)d_in[7];
    const float* bv    = (const float*)d_in[8];
    const float* g2    = (const float*)d_in[9];
    const float* beta2 = (const float*)d_in[10];
    const float* W1    = (const float*)d_in[11];
    const float* bm1   = (const float*)d_in[12];
    const float* W2    = (const float*)d_in[13];
    const float* bm2   = (const float*)d_in[14];
    float* out = (float*)d_out;

    char* p = (char*)d_ws;
    size_t off = 0;
    auto alloc = [&](size_t bytes) { char* r = p + off; off = align256(off + bytes); return r; };
    unsigned short* hb    = (unsigned short*)alloc((size_t)ROWS * DD * 2);
    unsigned short* qb    = (unsigned short*)alloc((size_t)BB * HH * SS * HD * 2);
    unsigned short* kb_   = (unsigned short*)alloc((size_t)BB * HH * SS * HD * 2);
    unsigned short* vTb   = (unsigned short*)alloc((size_t)BB * HH * HD * SS * 2);
    float*          out1  = (float*)        alloc((size_t)ROWS * DD * 4);
    unsigned short* h2b   = (unsigned short*)alloc((size_t)ROWS * DD * 2);
    unsigned short* mg    = (unsigned short*)alloc((size_t)ROWS * MLPD * 2);
    unsigned short* Wqb   = (unsigned short*)alloc((size_t)HH * HD * HD * 2);
    unsigned short* Wkb   = (unsigned short*)alloc((size_t)HH * HD * HD * 2);
    unsigned short* Wvb   = (unsigned short*)alloc((size_t)HH * HD * HD * 2);
    unsigned short* W1b   = (unsigned short*)alloc((size_t)MLPD * DD * 2);
    unsigned short* W2b   = (unsigned short*)alloc((size_t)DD * MLPD * 2);
    (void)ws_size;

    const int nW = HH * HD * HD;
    const int nM = MLPD * DD;
    cvt_bf16_kernel<<<(nW + 255) / 256, 256, 0, stream>>>(Wq, Wqb, nW);
    cvt_bf16_kernel<<<(nW + 255) / 256, 256, 0, stream>>>(Wk, Wkb, nW);
    cvt_bf16_kernel<<<(nW + 255) / 256, 256, 0, stream>>>(Wv, Wvb, nW);
    cvt_bf16_kernel<<<(nM + 255) / 256, 256, 0, stream>>>(W1, W1b, nM);
    cvt_bf16_kernel<<<(nM + 255) / 256, 256, 0, stream>>>(W2, W2b, nM);

    ln_kernel<<<ROWS, 256, 0, stream>>>(x, g1, beta1, hb);

    qkv_kernel<<<BB * HH, 256, 0, stream>>>(hb, Wqb, bq, Wkb, bk, Wvb, bv, qb, kb_, vTb);

    attn_kernel<<<BB * HH * (SS / 16) / 4, 128, 0, stream>>>(qb, kb_, vTb, x, out1);

    ln_kernel<<<ROWS, 256, 0, stream>>>(out1, g2, beta2, h2b);

    mlp1_kernel<<<(ROWS / 64) * (MLPD / 256), 256, 0, stream>>>(h2b, W1b, bm1, mg);

    mlp2_kernel<<<(ROWS / 64) * (DD / 256), 256, 0, stream>>>(mg, W2b, bm2, out1, out);
}
